// MoE_28097676051036
// MI455X (gfx1250) — matrix operand order repaired and checked by reading, not compiled, not run on hardware
//
#include <hip/hip_runtime.h>
#include <hip/hip_bf16.h>

// ---------------------------------------------------------------------------
// MoE forward (top-2 of 8 experts), D=512, H=1024, N=8192 tokens.
// Routed (sparse) execution = 1/4 of the reference's dense FLOPs.
// Main FLOPs: V_WMMA_F32_16X16X32_BF16 (f32 accumulation).
// x-tile gather: GLOBAL_LOAD_ASYNC_TO_LDS_B128 + s_wait_asynccnt.
// ---------------------------------------------------------------------------

#define NTOK   8192
#define DIM    512
#define HID    1024
#define NEXP   8
#define MT     32                      // tokens per FFN block

typedef unsigned short u16;
typedef unsigned short u16x4  __attribute__((ext_vector_type(4)));
typedef unsigned short u16x8  __attribute__((ext_vector_type(8)));
typedef unsigned short u16x16 __attribute__((ext_vector_type(16)));
typedef __bf16         bf16x16 __attribute__((ext_vector_type(16)));
typedef float          f32x8  __attribute__((ext_vector_type(8)));

union FragU {
    u16x16  u;
    u16x8   h[2];
    bf16x16 b;
};

__device__ __forceinline__ u16 f2bf(float f) {
    __bf16 b = (__bf16)f;              // hardware v_cvt_*bf16_f32 (RNE)
    return __builtin_bit_cast(u16, b);
}

// ---------------------------------------------------------------------------
// pack_x: x [N,D] f32 -> bf16 row-major (async-copy friendly in FFN).
// ---------------------------------------------------------------------------
__global__ __launch_bounds__(256) void pack_x_kernel(const float* __restrict__ x,
                                                     u16* __restrict__ xb) {
    const size_t groups = (size_t)NTOK * DIM / 8;
    for (size_t g = (size_t)blockIdx.x * blockDim.x + threadIdx.x;
         g < groups; g += (size_t)gridDim.x * blockDim.x) {
        size_t base = g * 8;
        float4 v0 = *(const float4*)(x + base);
        float4 v1 = *(const float4*)(x + base + 4);
        u16x8 o;
        o[0] = f2bf(v0.x); o[1] = f2bf(v0.y); o[2] = f2bf(v0.z); o[3] = f2bf(v0.w);
        o[4] = f2bf(v1.x); o[5] = f2bf(v1.y); o[6] = f2bf(v1.z); o[7] = f2bf(v1.w);
        *(u16x8*)(xb + base) = o;
    }
}

// ---------------------------------------------------------------------------
// Pack w1 [E,D,H] (f32) into per-fragment bf16 B-operand layout.
// Fragment = 32(K) x 16(N): lanes 0-15 hold K=0..15 (j==K), lanes 16-31 hold
// K=16..31 (j==K-16), N = lane&15.  Fragment id = nb*KB + kb, 512 elems.
// ---------------------------------------------------------------------------
__global__ __launch_bounds__(256) void pack_w1_kernel(const float* __restrict__ w,
                                                      u16* __restrict__ wp) {
    const size_t total = (size_t)NEXP * DIM * HID;
    for (size_t idx = (size_t)blockIdx.x * blockDim.x + threadIdx.x;
         idx < total; idx += (size_t)gridDim.x * blockDim.x) {
        int e   = (int)(idx >> 19);
        int rem = (int)(idx & 524287);
        int k   = rem >> 10;                 // D index (K of GEMM1)
        int n   = rem & 1023;                // H index (N of GEMM1)
        int kb = k >> 5, kk = k & 31;
        int nb = n >> 4, nn = n & 15;
        int lanei = (kk < 16) ? nn : (16 + nn);
        int j     = kk & 15;
        size_t dst = ((size_t)e << 19) + ((size_t)(nb * 16 + kb) << 9)
                   + (size_t)lanei * 16 + j;                 // KB = 16
        wp[dst] = f2bf(w[idx]);
    }
}

// w2 [E,H,D] : K = H (KB = 32), N = D.
__global__ __launch_bounds__(256) void pack_w2_kernel(const float* __restrict__ w,
                                                      u16* __restrict__ wp) {
    const size_t total = (size_t)NEXP * HID * DIM;
    for (size_t idx = (size_t)blockIdx.x * blockDim.x + threadIdx.x;
         idx < total; idx += (size_t)gridDim.x * blockDim.x) {
        int e   = (int)(idx >> 19);
        int rem = (int)(idx & 524287);
        int k   = rem >> 9;                  // H index
        int n   = rem & 511;                 // D index
        int kb = k >> 5, kk = k & 31;
        int nb = n >> 4, nn = n & 15;
        int lanei = (kk < 16) ? nn : (16 + nn);
        int j     = kk & 15;
        size_t dst = ((size_t)e << 19) + ((size_t)(nb * 32 + kb) << 9)
                   + (size_t)lanei * 16 + j;                 // KB = 32
        wp[dst] = f2bf(w[idx]);
    }
}

// ---------------------------------------------------------------------------
// Gating: one wave per token; softmax + top-2 + atomic compaction + loss stats.
// ---------------------------------------------------------------------------
__global__ __launch_bounds__(256) void gate_kernel(const float* __restrict__ x,
                                                   const float* __restrict__ wg,
                                                   int*   __restrict__ cnt,
                                                   int*   __restrict__ etok,
                                                   float* __restrict__ egate,
                                                   float* __restrict__ psum,
                                                   float* __restrict__ zsum) {
    const int lane = threadIdx.x & 31;
    const int wave = threadIdx.x >> 5;
    const int tok  = blockIdx.x * 8 + wave;
    if (tok >= NTOK) return;

    float a[NEXP];
#pragma unroll
    for (int e = 0; e < NEXP; ++e) a[e] = 0.f;

    const float* xr = x + (size_t)tok * DIM;
    for (int d = lane; d < DIM; d += 32) {
        float xv = xr[d];
#pragma unroll
        for (int e = 0; e < NEXP; ++e)
            a[e] = fmaf(xv, wg[e * DIM + d], a[e]);
    }
#pragma unroll
    for (int e = 0; e < NEXP; ++e) {
#pragma unroll
        for (int off = 16; off > 0; off >>= 1)
            a[e] += __shfl_xor(a[e], off, 32);
    }

    if (lane == 0) {
        float mx = a[0];
#pragma unroll
        for (int e = 1; e < NEXP; ++e) mx = fmaxf(mx, a[e]);
        float p[NEXP];
        float s = 0.f;
#pragma unroll
        for (int e = 0; e < NEXP; ++e) { p[e] = __expf(a[e] - mx); s += p[e]; }
        float inv = 1.f / s;
        float lse = mx + __logf(s);

        int i1 = 0;
#pragma unroll
        for (int e = 1; e < NEXP; ++e) if (p[e] > p[i1]) i1 = e;
        int i2 = (i1 == 0) ? 1 : 0;
#pragma unroll
        for (int e = 0; e < NEXP; ++e) if (e != i1 && p[e] > p[i2]) i2 = e;

        float p1 = p[i1] * inv, p2 = p[i2] * inv;
        float dn = 1.f / (p1 + p2 + 1e-6f);

        int pos1 = atomicAdd(&cnt[i1], 1);
        etok[i1 * NTOK + pos1]  = tok;
        egate[i1 * NTOK + pos1] = p1 * dn;
        int pos2 = atomicAdd(&cnt[i2], 1);
        etok[i2 * NTOK + pos2]  = tok;
        egate[i2 * NTOK + pos2] = p2 * dn;

#pragma unroll
        for (int e = 0; e < NEXP; ++e) atomicAdd(&psum[e], p[e] * inv);
        atomicAdd(zsum, lse * lse);
    }
}

// Loss: CVLOSS = 0, so loss = 0.01*switch + 0.001*zloss.
__global__ void loss_kernel(const float* __restrict__ psum,
                            const int*   __restrict__ cnt,
                            const float* __restrict__ zsum,
                            float* __restrict__ out_loss) {
    if (threadIdx.x == 0 && blockIdx.x == 0) {
        float ps = 0.f, fs = 0.f;
        for (int e = 0; e < NEXP; ++e) { ps += psum[e]; fs += (float)cnt[e]; }
        float sw = 0.f;
        for (int e = 0; e < NEXP; ++e)
            sw += (psum[e] / ps) * ((float)cnt[e] / fs);
        sw *= (float)NEXP;
        float zl = zsum[0] / (float)NTOK;
        out_loss[0] = 0.01f * sw + 0.001f * zl;
    }
}

// ---------------------------------------------------------------------------
// Fused routed FFN. Block = 256 thr = 8 waves, one 32-token tile of one
// expert. x rows (bf16) gathered via GLOBAL_LOAD_ASYNC_TO_LDS_B128.
// Per 128-col H chunk: h = relu(x@w1+b1) (WMMA) -> bf16 LDS -> y += h@w2
// (WMMA, 32x64 f32 acc per wave). Epilogue: gate*(acc+b2) atomic-added to y.
// ---------------------------------------------------------------------------
__global__ __launch_bounds__(256) void ffn_kernel(const u16*  __restrict__ xb,
                                                  const float* __restrict__ b1,
                                                  const float* __restrict__ b2,
                                                  const u16*  __restrict__ w1p,
                                                  const u16*  __restrict__ w2p,
                                                  const int*  __restrict__ cnt,
                                                  const int*  __restrict__ etok,
                                                  const float* __restrict__ egate,
                                                  float* __restrict__ out) {
    const int e     = blockIdx.y;
    const int tile  = blockIdx.x;
    const int count = cnt[e];
    const int start = tile * MT;
    if (start >= count) return;
    const int nvalid = min(MT, count - start);

    __shared__ u16   xl[MT][DIM + 8];     // x tile, bf16 (pitch 520 = 65x16B)
    __shared__ u16   hl[MT][128 + 8];     // h chunk, bf16 (pitch 136 = 17x16B)
    __shared__ int   stok[MT];
    __shared__ float sgate[MT];

    const int t = threadIdx.x;
    if (t < MT) {
        int tok; float g;
        if (t < nvalid) {
            tok = etok[e * NTOK + start + t];
            g   = egate[e * NTOK + start + t];
        } else {
            tok = etok[e * NTOK + start];  // padded row: valid address, zero gate
            g   = 0.f;
        }
        stok[t] = tok;
        sgate[t] = g;
    }
    __syncthreads();

    // ---- async gather: 32 bf16 rows (1KB each) -> LDS, 8 threads per row.
    {
        const int row = t >> 3;            // 0..31
        const int seg = t & 7;             // 8 x 128B segments per row
        const unsigned lds_base =
            (unsigned)(size_t)&xl[row][seg * 64];          // byte offset in LDS
        const unsigned long long ga =
            (unsigned long long)(const void*)(xb + (size_t)stok[row] * DIM + seg * 64);
#pragma unroll
        for (int q = 0; q < 8; ++q) {      // 8 x 16B per thread
            unsigned lo = lds_base + q * 16;
            unsigned long long g = ga + q * 16;
            asm volatile("global_load_async_to_lds_b128 %0, %1, off"
                         :: "v"(lo), "v"(g) : "memory");
        }
    }
    asm volatile("s_wait_asynccnt 0x0" ::: "memory");
    __syncthreads();

    const int lane = t & 31;
    const int wave = t >> 5;
    const int hi   = lane >> 4;
    const int m16  = lane & 15;

    f32x8 acc[2][4];                       // 32 rows x 64 cols per wave
#pragma unroll
    for (int mt = 0; mt < 2; ++mt)
#pragma unroll
        for (int i = 0; i < 4; ++i)
#pragma unroll
            for (int r = 0; r < 8; ++r) acc[mt][i][r] = 0.f;

    const u16* w1e = w1p + ((size_t)e << 19);
    const u16* w2e = w2p + ((size_t)e << 19);

    for (int chunk = 0; chunk < 8; ++chunk) {          // H in chunks of 128
        // ---- GEMM1: h[32, chunk*128 + wave*16 .. +16)
        f32x8 c1[2];
#pragma unroll
        for (int mt = 0; mt < 2; ++mt)
#pragma unroll
            for (int r = 0; r < 8; ++r) c1[mt][r] = 0.f;

        const int nb = chunk * 8 + wave;               // H N-tile (KB=16)
        const u16* bb = w1e + ((size_t)(nb * 16) << 9) + (size_t)lane * 16;
        if (chunk < 7) {                               // prefetch next chunk
            __builtin_prefetch(bb + ((size_t)(8 * 16) << 9), 0, 0);
            __builtin_prefetch(w2e + ((size_t)(wave * 128 + (chunk + 1) * 4) << 9)
                               + (size_t)lane * 16, 0, 0);
        }
#pragma unroll 4
        for (int kb = 0; kb < 16; ++kb) {              // K = 512 in steps of 32
            FragU b;
            const u16x8* bp = (const u16x8*)(bb + ((size_t)kb << 9));
            b.h[0] = bp[0];
            b.h[1] = bp[1];
#pragma unroll
            for (int mt = 0; mt < 2; ++mt) {
                FragU a;
                a.h[0] = *(const u16x8*)&xl[mt * 16 + m16][kb * 32 + hi * 16];
                a.h[1] = *(const u16x8*)&xl[mt * 16 + m16][kb * 32 + hi * 16 + 8];
                c1[mt] = __builtin_amdgcn_wmma_f32_16x16x32_bf16(
                    false, a.b, false, b.b, (short)0, c1[mt], false, false);
            }
        }

        const float b1v = b1[(size_t)e * HID + chunk * 128 + wave * 16 + m16];
        __syncthreads();   // previous chunk's hl fully consumed
#pragma unroll
        for (int mt = 0; mt < 2; ++mt)
#pragma unroll
            for (int r = 0; r < 8; ++r) {
                float v = c1[mt][r] + b1v;
                v = v > 0.f ? v : 0.f;                 // ReLU
                hl[mt * 16 + hi * 8 + r][wave * 16 + m16] = f2bf(v);
            }
        __syncthreads();

        // ---- GEMM2: y[32, wave*64 .. +64] += h_chunk @ w2_chunk (KB=32)
#pragma unroll
        for (int nt = 0; nt < 4; ++nt) {
            const int nb2 = wave * 4 + nt;
            const u16* b2b = w2e + ((size_t)(nb2 * 32 + chunk * 4) << 9)
                           + (size_t)lane * 16;
#pragma unroll
            for (int kb2 = 0; kb2 < 4; ++kb2) {        // 128 = 4 x 32
                FragU b;
                const u16x8* bp = (const u16x8*)(b2b + ((size_t)kb2 << 9));
                b.h[0] = bp[0];
                b.h[1] = bp[1];
#pragma unroll
                for (int mt = 0; mt < 2; ++mt) {
                    FragU a;
                    a.h[0] = *(const u16x8*)&hl[mt * 16 + m16][kb2 * 32 + hi * 16];
                    a.h[1] = *(const u16x8*)&hl[mt * 16 + m16][kb2 * 32 + hi * 16 + 8];
                    acc[mt][nt] = __builtin_amdgcn_wmma_f32_16x16x32_bf16(
                        false, a.b, false, b.b, (short)0, acc[mt][nt], false, false);
                }
            }
        }
    }

    // ---- epilogue: gate * (acc + b2) -> atomic add into y
#pragma unroll
    for (int nt = 0; nt < 4; ++nt) {
        const int col  = wave * 64 + nt * 16 + m16;
        const float bv = b2[(size_t)e * DIM + col];
#pragma unroll
        for (int mt = 0; mt < 2; ++mt)
#pragma unroll
            for (int r = 0; r < 8; ++r) {
                const int row = mt * 16 + hi * 8 + r;
                if (row < nvalid) {
                    float val = sgate[row] * (acc[mt][nt][r] + bv);
                    atomicAdd(out + (size_t)stok[row] * DIM + col, val);
                }
            }
    }
}

// ---------------------------------------------------------------------------
extern "C" void kernel_launch(void* const* d_in, const int* in_sizes, int n_in,
                              void* d_out, int out_size, void* d_ws, size_t ws_size,
                              hipStream_t stream) {
    const float* x  = (const float*)d_in[0];   // [4,2048,512]
    const float* wg = (const float*)d_in[1];   // [8,512]
    const float* w1 = (const float*)d_in[2];   // [8,512,1024]
    const float* b1 = (const float*)d_in[3];   // [8,1024]
    const float* w2 = (const float*)d_in[4];   // [8,1024,512]
    const float* b2 = (const float*)d_in[5];   // [8,512]
    float* out = (float*)d_out;                // [N*D] y, then [1] loss

    // Workspace layout
    char* p = (char*)d_ws;
    int*   cnt   = (int*)  (p + 0);                         // 8 * 4B
    float* psum  = (float*)(p + 32);                        // 8 * 4B
    float* zsum  = (float*)(p + 64);                        // 4B
    int*   etok  = (int*)  (p + 128);                       // 256KB
    float* egate = (float*)(p + 128 + 262144);              // 256KB
    u16*   xbf   = (u16*)  (p + 128 + 2 * 262144);          // 8MB bf16 x
    u16*   w1p   = (u16*)  (p + 128 + 2 * 262144 + 8388608);        // 8MB
    u16*   w2p   = (u16*)  (p + 128 + 2 * 262144 + 2 * 8388608);    // 8MB

    // Zero y + loss, and the gating accumulators.
    hipMemsetAsync(d_out, 0, (size_t)out_size * sizeof(float), stream);
    hipMemsetAsync(d_ws, 0, 128, stream);

    // f32 -> bf16 conversion / fragment packing.
    pack_x_kernel <<<2048, 256, 0, stream>>>(x, xbf);
    pack_w1_kernel<<<2048, 256, 0, stream>>>(w1, w1p);
    pack_w2_kernel<<<2048, 256, 0, stream>>>(w2, w2p);

    // Gating: 8 tokens (waves) per block.
    gate_kernel<<<NTOK / 8, 256, 0, stream>>>(x, wg, cnt, etok, egate, psum, zsum);

    // Loss scalar.
    loss_kernel<<<1, 32, 0, stream>>>(psum, cnt, zsum, out + (size_t)NTOK * DIM);

    // Routed FFN: grid covers worst case (all tokens on one expert).
    dim3 grid(NTOK / MT, NEXP);
    ffn_kernel<<<grid, 256, 0, stream>>>(xbf, b1, b2, w1p, w2p, cnt, etok, egate, out);
}
